// MultiscaleGlobalDynamicDomainFilters_19542101196963
// MI455X (gfx1250) — matrix operand order repaired and checked by reading, not compiled, not run on hardware
//
#include <hip/hip_runtime.h>
#include <hip/hip_bf16.h>

typedef __attribute__((ext_vector_type(16))) _Float16 v16h;
typedef __attribute__((ext_vector_type(8)))  _Float16 v8h;
typedef __attribute__((ext_vector_type(4)))  _Float16 v4h;
typedef __attribute__((ext_vector_type(2)))  _Float16 v2h;
typedef __attribute__((ext_vector_type(8)))  float    v8f;
typedef __attribute__((ext_vector_type(4)))  float    v4f;
typedef __attribute__((ext_vector_type(4)))  unsigned int u32x4;
typedef __attribute__((ext_vector_type(4)))  int          i32x4;
typedef __attribute__((ext_vector_type(8)))  int          i32x8;

// ---------------------------------------------------------------------------
// Pack a row-major f32 matrix W[M x K] into CDNA5 WMMA A-fragment layout
// (16x32 f16 tiles, contiguous k order shared with the B staging): element (mtile, kstep, lane, i) holds
//   W[mtile*16 + (lane&15)][kstep*32 + (lane>>4)*16 + i]
// so each lane of a wave can load its v16h operand with one contiguous 32B read.
// ---------------------------------------------------------------------------
__global__ __launch_bounds__(256) void packA_kernel(const float* __restrict__ Wsrc,
                                                    _Float16* __restrict__ dst,
                                                    int Ksteps, int K, int total) {
    int idx = blockIdx.x * 256 + threadIdx.x;
    if (idx >= total) return;
    int i     = idx & 15;
    int lane  = (idx >> 4) & 31;
    int t     = idx >> 9;              // mtile*Ksteps + kstep
    int ks    = t % Ksteps;
    int mtile = t / Ksteps;
    int row   = mtile * 16 + (lane & 15);
    int col   = ks * 32 + ((lane >> 4) * 16) + i;
    dst[idx] = (_Float16)Wsrc[(size_t)row * K + col];
}

// ---------------------------------------------------------------------------
// Global average pool: xm[n][c] = mean_{hw} x[n][c][hw].  One block per (n,c).
// ---------------------------------------------------------------------------
__global__ __launch_bounds__(256) void mean_kernel(const float* __restrict__ x,
                                                   float* __restrict__ xm) {
    __shared__ float red[256];
    const size_t base = (size_t)blockIdx.x * 4096;
    float s = 0.f;
    for (int i = 0; i < 16; ++i) s += x[base + threadIdx.x + i * 256];
    red[threadIdx.x] = s;
    __syncthreads();
    for (int st = 128; st > 0; st >>= 1) {
        if (threadIdx.x < st) red[threadIdx.x] += red[threadIdx.x + st];
        __syncthreads();
    }
    if (threadIdx.x == 0) xm[blockIdx.x] = red[0] * (1.f / 4096.f);
}

// ---------------------------------------------------------------------------
// g[n][o] = relu( sum_c xm[n][c] * cw[o][c] + conv_b[o] )   (16x128, K=512)
// ---------------------------------------------------------------------------
__global__ __launch_bounds__(128) void g_kernel(const float* __restrict__ xm,
                                                const float* __restrict__ cw,
                                                const float* __restrict__ cb,
                                                float* __restrict__ g) {
    int n = blockIdx.x, o = threadIdx.x;
    const float* xr = xm + n * 512;
    const float* wr = cw + o * 512;
    float acc = cb[o];
    for (int c = 0; c < 512; ++c) acc += xr[c] * wr[c];
    g[n * 128 + o] = fmaxf(acc, 0.f);
}

// ---------------------------------------------------------------------------
// Dynamic depthwise kernels (tiny VALU work).
// ---------------------------------------------------------------------------
__global__ __launch_bounds__(128) void dynw_kernel(const float* __restrict__ g,
                                                   const float* __restrict__ ckw,
                                                   const float* __restrict__ ckb,
                                                   const float* __restrict__ ck2w,
                                                   const float* __restrict__ ck2b,
                                                   const float* __restrict__ wk,
                                                   const float* __restrict__ kb,
                                                   const float* __restrict__ wk2,
                                                   const float* __restrict__ kb2,
                                                   float* __restrict__ Wd,
                                                   float* __restrict__ W2) {
    int n = blockIdx.x, o = threadIdx.x;
    float gm = 0.f;
    for (int nn = 0; nn < 16; ++nn) gm += g[nn * 128 + o];
    gm *= (1.f / 16.f);
    float gv = g[n * 128 + o];
    float a0 = ckw[0], a1 = ckw[1], b0 = ckb[0];
    float c0 = ck2w[0], c1 = ck2w[1], d0 = ck2b[0];
    int idx = n * 128 + o;
    for (int j = 0; j < 25; ++j) {
        float Ki = gv * wk[j] + kb[j];
        float Kd = gm * wk[j] + kb[j];
        Wd[idx * 25 + j] = fmaxf(a0 * Ki + a1 * Kd + b0, 0.f);
    }
    for (int j = 0; j < 9; ++j) {
        float Ki = gv * wk2[j] + kb2[j];
        float Kd = gm * wk2[j] + kb2[j];
        W2[idx * 9 + j] = fmaxf(c0 * Ki + c1 * Kd + d0, 0.f);
    }
}

// ---------------------------------------------------------------------------
// WMMA GEMM: out[n][m][p] = (relu?)(sum_k A[m][k] * src[n][k][p] + bias[m])
//  - A pre-packed in WMMA A-fragment layout (f16), contiguous 32B per lane
//  - src staged through LDS in B-fragment layout (double buffered); each
//    thread stages a K-pair x 4 columns so LDS stores are packed b32
//  - block = 8 waves; wave owns MT M-tiles x 64 N columns (MT=2 for the big
//    fuse GEMM -> 8 WMMA per 8 ds_load_b128 per K-step)
// ---------------------------------------------------------------------------
template <bool SRC_F16, bool RELU, int MT>
__global__ __launch_bounds__(256) void gemm_kernel(const void* __restrict__ srcv,
                                                   const _Float16* __restrict__ packedA,
                                                   const float* __restrict__ bias,
                                                   float* __restrict__ out,
                                                   int K, int Mtot) {
    const int Ksteps = K >> 5;
    const int n      = blockIdx.y;
    const int pbase  = blockIdx.x * 64;
    const int wave   = threadIdx.x >> 5;
    const int lane   = threadIdx.x & 31;

    // B fragments for 4 N-subtiles, double buffered: [buf][subtile][lane][i]
    __shared__ __align__(32) _Float16 ldsB[2][4][32][16];

    // Loader mapping: thread stages K-pair (k0,k0+1) x 4 columns.
    const int kp    = threadIdx.x >> 4;      // 0..15
    const int k0    = kp * 2;
    const int tcol  = threadIdx.x & 15;
    const int p0    = tcol * 4;
    const int sidx  = tcol >> 2;             // N-subtile
    const int lhalf = (k0 >= 16) ? 16 : 0;
    const int i0    = k0 & 15;               // even -> b32-aligned pair store

    auto stage = [&](int ks, int buf) {
        const size_t r0 = ((size_t)n * K + (size_t)ks * 32 + k0) * 4096 + pbase + p0;
        v2h pr[4];
        if (SRC_F16) {
            const _Float16* s16 = (const _Float16*)srcv;
            v4h u = *(const v4h*)(s16 + r0);
            v4h w = *(const v4h*)(s16 + r0 + 4096);
            for (int e = 0; e < 4; ++e) { pr[e][0] = u[e]; pr[e][1] = w[e]; }
        } else {
            const float* s32 = (const float*)srcv;
            v4f u = *(const v4f*)(s32 + r0);
            v4f w = *(const v4f*)(s32 + r0 + 4096);
            for (int e = 0; e < 4; ++e) {
                pr[e][0] = (_Float16)u[e];
                pr[e][1] = (_Float16)w[e];
            }
        }
        for (int e = 0; e < 4; ++e) {
            int p = p0 + e;
            *(v2h*)&ldsB[buf][sidx][lhalf + (p & 15)][i0] = pr[e];
        }
    };

    v8f acc[MT][4];
#pragma unroll
    for (int t = 0; t < MT; ++t)
#pragma unroll
        for (int s = 0; s < 4; ++s) acc[t][s] = (v8f){};

    stage(0, 0);
    __syncthreads();

    for (int ks = 0; ks < Ksteps; ++ks) {
        const int cur = ks & 1;
        if (ks + 1 < Ksteps) stage(ks + 1, cur ^ 1);

        v16h b0 = *(const v16h*)&ldsB[cur][0][lane][0];
        v16h b1 = *(const v16h*)&ldsB[cur][1][lane][0];
        v16h b2 = *(const v16h*)&ldsB[cur][2][lane][0];
        v16h b3 = *(const v16h*)&ldsB[cur][3][lane][0];

#pragma unroll
        for (int t = 0; t < MT; ++t) {
            const int mtile = blockIdx.z * (8 * MT) + t * 8 + wave;
            v16h a = *(const v16h*)(packedA + (((size_t)mtile * Ksteps + ks) * 32 + lane) * 16);
            acc[t][0] = __builtin_amdgcn_wmma_f32_16x16x32_f16(false, a, false, b0, (short)0, acc[t][0], false, false);
            acc[t][1] = __builtin_amdgcn_wmma_f32_16x16x32_f16(false, a, false, b1, (short)0, acc[t][1], false, false);
            acc[t][2] = __builtin_amdgcn_wmma_f32_16x16x32_f16(false, a, false, b2, (short)0, acc[t][2], false, false);
            acc[t][3] = __builtin_amdgcn_wmma_f32_16x16x32_f16(false, a, false, b3, (short)0, acc[t][3], false, false);
        }
        __syncthreads();
    }

    // C/D layout: VGPR r -> row r (lanes 0-15) / row r+8 (lanes 16-31), col = lane&15
    const int rowoff = (lane >= 16) ? 8 : 0;
    const int col    = lane & 15;
#pragma unroll
    for (int t = 0; t < MT; ++t) {
        const int m0 = (blockIdx.z * (8 * MT) + t * 8 + wave) * 16;
#pragma unroll
        for (int r = 0; r < 8; ++r) {
            int    m  = m0 + r + rowoff;
            float  bv = bias[m];
            size_t ob = ((size_t)n * Mtot + m) * 4096 + pbase + col;
            float o0 = acc[t][0][r] + bv;
            float o1 = acc[t][1][r] + bv;
            float o2 = acc[t][2][r] + bv;
            float o3 = acc[t][3][r] + bv;
            if (RELU) {
                o0 = fmaxf(o0, 0.f); o1 = fmaxf(o1, 0.f);
                o2 = fmaxf(o2, 0.f); o3 = fmaxf(o3, 0.f);
            }
            out[ob]      = o0;
            out[ob + 16] = o1;
            out[ob + 32] = o2;
            out[ob + 48] = o3;
        }
    }
}

// ---------------------------------------------------------------------------
// Depthwise 5x5 (pad 2) + dilated 3x3 (dil 2, pad 2) with per-(n,c) dynamic
// kernels. The halo tile is staged by the Tensor Data Mover: a 2D tile of
// 64-float rows DMA'd into LDS with a 68-float pitch via TDM LDS padding
// (pad_interval = 64 DWORDs, pad_amount = 4 DWORDs). Bottom-edge rows are
// zero-filled by tensor-dim OOB; top edge is clamped with an LDS offset.
// Dynamic shared memory only => TDM lds_addr is a plain offset from 0.
// ---------------------------------------------------------------------------
__global__ __launch_bounds__(256) void depthwise_kernel(const float* __restrict__ f,
                                                        const float* __restrict__ Wd,
                                                        const float* __restrict__ W2,
                                                        const float* __restrict__ adap_bias,
                                                        const float* __restrict__ atrous_bias,
                                                        _Float16* __restrict__ ybuf) {
    extern __shared__ float smem[];           // [20*68] tile | [25] w5 | [9] w3
    float (*tile)[68] = (float(*)[68])smem;
    float* w5 = smem + 20 * 68;
    float* w3 = w5 + 25;

    const int bx = blockIdx.x;
    const int hq = bx & 3;
    const int c  = (bx >> 2) & 127;
    const int n  = bx >> 9;
    const int h0 = hq * 16;
    const int nc = n * 128 + c;
    const float* fp = f + (size_t)nc * 4096;

    // Pre-zero tile (halo cols + top edge rows) and load dynamic weights.
    for (int e = threadIdx.x; e < 20 * 68; e += 256) smem[e] = 0.f;
    if (threadIdx.x < 25) w5[threadIdx.x] = Wd[nc * 25 + threadIdx.x];
    if (threadIdx.x >= 32 && threadIdx.x < 41) w3[threadIdx.x - 32] = W2[nc * 9 + (threadIdx.x - 32)];
    __syncthreads();

    // Wave 0 issues the TDM tile load (uniform SGPR descriptors).
    if (threadIdx.x < 32) {
        const int r0       = (h0 - 2 < 0) ? 0 : (h0 - 2);  // clamped first row
        const int skipped  = r0 - (h0 - 2);                // 0..2 top rows skipped
        const unsigned rows   = 20u - (unsigned)skipped;   // tile_dim1
        const unsigned tdim1  = 64u - (unsigned)r0;        // rows available from tile start
        const unsigned ldsoff = (unsigned)((skipped * 68 + 2) * 4);
        const unsigned long long gaddr = (unsigned long long)(uintptr_t)(fp + r0 * 64);

        u32x4 g0;
        g0[0] = 1u;                                        // count=1, user descriptor
        g0[1] = ldsoff;                                    // lds_addr
        g0[2] = (unsigned)(gaddr & 0xFFFFFFFFu);           // global_addr[31:0]
        g0[3] = (unsigned)((gaddr >> 32) & 0x1FFFFFFu)     // global_addr[56:32]
              | 0x80000000u;                               // type=2 ("image")
        i32x8 g1;
        g1[0] = (int)(0x00020000u                          // data_size=4B
              | (1u << 20)                                 // pad_enable
              | (5u << 22)                                 // pad_interval: 64 DWORDs
              | (3u << 25));                               // pad_amount: 4 DWORDs
        g1[1] = (int)(64u << 16);                          // tensor_dim0 = 64 (lo16)
        g1[2] = (int)((tdim1 & 0xFFFFu) << 16);            // tensor_dim1 lo16
        g1[3] = (int)((tdim1 >> 16) | (64u << 16));        // tensor_dim1 hi16 | tile_dim0=64
        g1[4] = (int)rows;                                 // tile_dim1 (tile_dim2=0 -> 2D)
        g1[5] = 64;                                        // tensor_dim0_stride = 64
        g1[6] = 0;
        g1[7] = 0;
        i32x4 z4 = {};
#if __clang_major__ >= 23
        i32x8 z8 = {};
        __builtin_amdgcn_tensor_load_to_lds(g0, g1, z4, z4, z8, 0);
#else
        __builtin_amdgcn_tensor_load_to_lds(g0, g1, z4, z4, 0);
#endif
        __builtin_amdgcn_s_wait_tensorcnt(0);
    }
    __syncthreads();

    const int   w   = threadIdx.x & 63;
    const int   rg  = threadIdx.x >> 6;
    const float ab  = adap_bias[c];
    const float atb = atrous_bias[c];
    _Float16* y1p = ybuf + ((size_t)n * 256 + c) * 4096;
    _Float16* y2p = ybuf + ((size_t)n * 256 + 128 + c) * 4096;

    for (int rr = 0; rr < 4; ++rr) {
        int lr = rg * 4 + rr;
        int h  = h0 + lr;
        float s1 = 0.f;
        for (int i = 0; i < 5; ++i)
            for (int j = 0; j < 5; ++j)
                s1 += w5[i * 5 + j] * tile[lr + i][w + j];
        float s2 = 0.f;
        for (int i = 0; i < 3; ++i)
            for (int j = 0; j < 3; ++j)
                s2 += w3[i * 3 + j] * tile[lr + 2 * i][w + 2 * j];
        y1p[h * 64 + w] = (_Float16)(s1 + ab);
        y2p[h * 64 + w] = (_Float16)(s2 + atb);
    }
}

// ---------------------------------------------------------------------------
extern "C" void kernel_launch(void* const* d_in, const int* in_sizes, int n_in,
                              void* d_out, int out_size, void* d_ws, size_t ws_size,
                              hipStream_t stream) {
    (void)in_sizes; (void)n_in; (void)out_size; (void)ws_size;

    const float* x              = (const float*)d_in[0];
    const float* conv_w         = (const float*)d_in[1];
    const float* conv_b         = (const float*)d_in[2];
    const float* conv_k_w       = (const float*)d_in[3];
    const float* conv_k_b       = (const float*)d_in[4];
    const float* conv_k2_w      = (const float*)d_in[5];
    const float* conv_k2_b      = (const float*)d_in[6];
    const float* conv_kernel_w  = (const float*)d_in[7];
    const float* conv_kernel_b  = (const float*)d_in[8];
    const float* conv_kernel2_w = (const float*)d_in[9];
    const float* conv_kernel2_b = (const float*)d_in[10];
    const float* adap_bias      = (const float*)d_in[11];
    const float* atrous_bias    = (const float*)d_in[12];
    const float* fuse_w         = (const float*)d_in[13];
    const float* fuse_b         = (const float*)d_in[14];
    float* out = (float*)d_out;

    char* ws = (char*)d_ws;
    size_t off = 0;
    auto take = [&](size_t bytes) -> char* {
        char* p = ws + off;
        off += (bytes + 255) & ~(size_t)255;
        return p;
    };
    float*    f_buf = (float*)take((size_t)16 * 128 * 4096 * 4);    // f, f32
    _Float16* ybuf  = (_Float16*)take((size_t)16 * 256 * 4096 * 2); // y, f16
    _Float16* pA_f  = (_Float16*)take((size_t)8 * 16 * 512 * 2);    // packed conv_w
    _Float16* pA_u  = (_Float16*)take((size_t)32 * 8 * 512 * 2);    // packed fuse_w
    float*    xm    = (float*)take((size_t)16 * 512 * 4);
    float*    gb    = (float*)take((size_t)16 * 128 * 4);
    float*    WdB   = (float*)take((size_t)2048 * 25 * 4);
    float*    W2B   = (float*)take((size_t)2048 * 9 * 4);

    // Weight packing for WMMA (A fragments)
    packA_kernel<<<(8 * 16 * 512 + 255) / 256, 256, 0, stream>>>(conv_w, pA_f, 16, 512, 8 * 16 * 512);
    packA_kernel<<<(32 * 8 * 512 + 255) / 256, 256, 0, stream>>>(fuse_w, pA_u, 8, 256, 32 * 8 * 512);

    // Dynamic-kernel generation path (tiny)
    mean_kernel<<<16 * 512, 256, 0, stream>>>(x, xm);
    g_kernel<<<16, 128, 0, stream>>>(xm, conv_w, conv_b, gb);
    dynw_kernel<<<16, 128, 0, stream>>>(gb, conv_k_w, conv_k_b, conv_k2_w, conv_k2_b,
                                        conv_kernel_w, conv_kernel_b,
                                        conv_kernel2_w, conv_kernel2_b, WdB, W2B);

    // f = relu(conv_w @ x + conv_b)   (WMMA, f32 src, M=128, K=512, MT=1)
    dim3 gf(64, 16, 1);
    gemm_kernel<false, true, 1><<<gf, 256, 0, stream>>>(x, pA_f, conv_b, f_buf, 512, 128);

    // y1/y2 depthwise with dynamic kernels -> ybuf (f16, concatenated); TDM tile load
    const size_t dw_smem = (size_t)(20 * 68 + 25 + 9) * sizeof(float);
    depthwise_kernel<<<16 * 128 * 4, 256, dw_smem, stream>>>(f_buf, WdB, W2B, adap_bias, atrous_bias, ybuf);

    // out = fuse_w @ y + fuse_b       (WMMA, f16 src, M=512, K=256, MT=2)
    dim3 gu(64, 16, 2);
    gemm_kernel<true, false, 2><<<gu, 256, 0, stream>>>(ybuf, pA_u, fuse_b, out, 256, 512);
}
